// BasicTransformerBlock_78151224918121
// MI455X (gfx1250) — compile-verified, matrix operand order checked by reading, not run on hardware
//
#include <hip/hip_runtime.h>
#include <cmath>

// ---------------------------------------------------------------------------
// CDNA5 (gfx1250) WMMA transformer block.
// wave32; v_wmma_f32_16x16x32_f16 for all GEMMs + both attention matmuls,
// async global->LDS staging (ASYNCcnt) for GEMM A tiles, ds_swizzle softmax
// reductions, global_prefetch for streamed K/V.
// ---------------------------------------------------------------------------

#ifndef USE_ASYNC_LDS
#define USE_ASYNC_LDS 1   // global_load_async_to_lds_b128 / s_wait_asynccnt
#endif

typedef __attribute__((ext_vector_type(16))) _Float16 v16h;
typedef __attribute__((ext_vector_type(8)))  float    v8f;

union FragH {
  v16h v;
  _Float16 h[16];
  uint4 q[2];
};

__device__ __forceinline__ v8f wmma16(const FragH& a, const FragH& b, v8f c) {
  // D = A(16x32 f16) * B(32x16 f16) + C(16x16 f32)
  return __builtin_amdgcn_wmma_f32_16x16x32_f16(
      false, a.v, false, b.v, (short)0, c, false, false);
}

// lane' = lane ^ m within 32 lanes: ds_swizzle group-of-32, and=0x1f, xor=m<<10
#define SWZ_XOR(x, imm) \
  __int_as_float(__builtin_amdgcn_ds_swizzle(__float_as_int(x), (imm)))

__device__ __forceinline__ void async_cp_b128(unsigned lds_off, const void* g) {
#if USE_ASYNC_LDS
  asm volatile("global_load_async_to_lds_b128 %0, %1, off"
               :: "v"(lds_off), "v"(g) : "memory");
#else
  *(uint4*)(unsigned long long)0 = *(const uint4*)g;  // unused
#endif
}
__device__ __forceinline__ void wait_async() {
#if USE_ASYNC_LDS
  asm volatile("s_wait_asynccnt 0" ::: "memory");
#endif
}

#define BM 64
#define BN 64
#define BK 32
#define BKP 40   // LDS row pitch in halves: 80B -> 16B aligned, conflict-free

// ---------------------------------------------------------------------------
// Generic f16 WMMA GEMM:  out = A[M,K] * Bw[K,N]  (+bias) (+res)
// Double-buffered LDS; A tile via async-to-LDS, B tile reg-preload+transpose.
// Epilogues:
//   EPI=0 : store f16 row-major (+bias if non-null)          (FF1)
//   EPI=1 : store f16 head-split   [bg*8+h][Mpad][64] d<40   (Q,K proj)
//   EPI=2 : store f16 head-split-T [bg*8+h][64][Mpad]        (V proj)
//   EPI=3 : store f32 row-major + bias                       (cross Wo)
//   EPI=4 : store f32 row-major + bias + residual            (self Wo, FF2)
// ---------------------------------------------------------------------------
template <int EPI>
__global__ __launch_bounds__(256) void gemm_f16_wmma(
    const _Float16* __restrict__ A, const _Float16* __restrict__ Bw,
    const float* __restrict__ bias, const float* __restrict__ res,
    void* __restrict__ outp, int M, int N, int K, int TPB, int Mpad)
{
  __shared__ alignas(16) _Float16 As[2][BM][BKP];
  __shared__ alignas(16) _Float16 Bs[2][BN][BKP];
  const int tid  = threadIdx.x;
  const int lane = tid & 31, wid = tid >> 5;
  const int hl   = lane >> 4, ln = lane & 15;
  const int tileM = blockIdx.y * BM, tileN = blockIdx.x * BN;
  const int wm = wid & 3, wn = wid >> 2;     // wave -> (row tile, col-tile pair)
  v8f acc0 = {}, acc1 = {};

  // A stage: thread -> 16B of a 64x32 tile (row-clamped; OOB rows discarded
  // in the epilogue, clamping keeps async reads in-bounds).
  const int arow = tid >> 2, aseg = tid & 3;
  const int am   = (tileM + arow < M) ? (tileM + arow) : (M - 1);
  const _Float16* aptr = A + (size_t)am * K + aseg * 8;
  // B stage: thread -> 16B of a 32x64 tile, transposed into LDS.
  const int bk = tid & 31, bcs = tid >> 5;
  const _Float16* bptr = Bw + (size_t)bk * N + tileN + bcs * 8;

  const unsigned aoff = aseg * 8;  // halves within row

#if USE_ASYNC_LDS
  async_cp_b128((unsigned)(size_t)&As[0][arow][aoff], aptr);
#else
  *(uint4*)&As[0][arow][aoff] = *(const uint4*)aptr;
#endif
  {
    uint4 bv0 = *(const uint4*)bptr;
    const _Float16* bh = (const _Float16*)&bv0;
#pragma unroll
    for (int c = 0; c < 8; ++c) Bs[0][bcs * 8 + c][bk] = bh[c];
  }
  wait_async();
  __syncthreads();

  int cur = 0;
  for (int kb = 0; kb < K; kb += BK) {
    const int nxt = kb + BK;
    uint4 bv;
    if (nxt < K) {
#if USE_ASYNC_LDS
      async_cp_b128((unsigned)(size_t)&As[cur ^ 1][arow][aoff], aptr + nxt);
#else
      *(uint4*)&As[cur ^ 1][arow][aoff] = *(const uint4*)(aptr + nxt);
#endif
      bv = *(const uint4*)(bptr + (size_t)nxt * N);
    }

    FragH af, bf0, bf1;
    const int ar = wm * 16 + ln;
    af.q[0] = *(const uint4*)(&As[cur][ar][hl * 8]);
    af.q[1] = *(const uint4*)(&As[cur][ar][16 + hl * 8]);
    const int bc0 = wn * 32 + ln;
    bf0.q[0] = *(const uint4*)(&Bs[cur][bc0][hl * 8]);
    bf0.q[1] = *(const uint4*)(&Bs[cur][bc0][16 + hl * 8]);
    bf1.q[0] = *(const uint4*)(&Bs[cur][bc0 + 16][hl * 8]);
    bf1.q[1] = *(const uint4*)(&Bs[cur][bc0 + 16][16 + hl * 8]);
    acc0 = wmma16(af, bf0, acc0);
    acc1 = wmma16(af, bf1, acc1);

    if (nxt < K) {
      const _Float16* bh = (const _Float16*)&bv;
#pragma unroll
      for (int c = 0; c < 8; ++c) Bs[cur ^ 1][bcs * 8 + c][bk] = bh[c];
      wait_async();
      __syncthreads();
      cur ^= 1;
    }
  }

  v8f accs[2] = {acc0, acc1};
#pragma unroll
  for (int t = 0; t < 2; ++t) {
    const int n = tileN + wn * 32 + t * 16 + ln;
#pragma unroll
    for (int r = 0; r < 8; ++r) {
      const int m = tileM + wm * 16 + hl * 8 + r;
      if (m >= M) continue;
      float vv = accs[t][r];
      if (EPI == 0) {
        if (bias) vv += bias[n];
        ((_Float16*)outp)[(size_t)m * N + n] = (_Float16)vv;
      } else if (EPI == 1 || EPI == 2) {
        const int hh = n / 40, dd = n - hh * 40;
        const int bg = m / TPB, tok = m - bg * TPB;
        _Float16* o = (_Float16*)outp;
        if (EPI == 1)
          o[(((size_t)bg * 8 + hh) * Mpad + tok) * 64 + dd] = (_Float16)vv;
        else
          o[(((size_t)bg * 8 + hh) * 64 + dd) * (size_t)Mpad + tok] = (_Float16)vv;
      } else {
        vv += bias[n];
        if (EPI == 4) vv += res[(size_t)m * N + n];
        ((float*)outp)[(size_t)m * N + n] = vv;
      }
    }
  }
}

// ---------------------------------------------------------------------------
// Fused flash attention (WMMA, online softmax). One block = 128 query rows of
// one (batch,head) group; 8 waves x 16 rows. Q: [G,Nq,64], K: [G,Mkp,64],
// Vt: [G,64,Mkp] (d-major so P*V B-frags are contiguous). DH=40 zero-padded
// to 64 in Q; keys >= Mk masked to -inf. Output f16 [*,320] at col h*40.
// ---------------------------------------------------------------------------
__global__ __launch_bounds__(256) void flash_attn_wmma(
    const _Float16* __restrict__ Q, const _Float16* __restrict__ Kk,
    const _Float16* __restrict__ Vt, _Float16* __restrict__ out,
    int Nq, int Mk, int Mkp, float scale)
{
  __shared__ alignas(16) _Float16 Pl[8][16][BKP];  // per-wave P staging
  const int tid = threadIdx.x, lane = tid & 31, wid = tid >> 5;
  const int hl = lane >> 4, ln = lane & 15;
  const int g  = blockIdx.y;
  const int hh = g & 7, bg = g >> 3;
  const _Float16* q  = Q  + (size_t)g * Nq * 64;
  const _Float16* kk = Kk + (size_t)g * Mkp * 64;
  const _Float16* vt = Vt + (size_t)g * 64 * Mkp;
  const int qrow0 = blockIdx.x * 128 + wid * 16;

  FragH qa0, qa1;
  {
    const _Float16* qr = q + (size_t)(qrow0 + ln) * 64;
    qa0.q[0] = *(const uint4*)(qr + hl * 8);
    qa0.q[1] = *(const uint4*)(qr + 16 + hl * 8);
    qa1.q[0] = *(const uint4*)(qr + 32 + hl * 8);
    qa1.q[1] = *(const uint4*)(qr + 48 + hl * 8);
  }

  v8f o0 = {}, o1 = {}, o2 = {};
  float mr[8], lr[8];
#pragma unroll
  for (int r = 0; r < 8; ++r) { mr[r] = -__builtin_inff(); lr[r] = 0.f; }

  for (int j0 = 0; j0 < Mkp; j0 += 32) {
    if (j0 + 32 < Mkp) {  // prefetch next K/V tiles (global_prefetch_b8)
      __builtin_prefetch(kk + (size_t)(j0 + 32 + ln) * 64, 0, 0);
      __builtin_prefetch(vt + (size_t)ln * Mkp + j0 + 32, 0, 0);
    }
    FragH kb00, kb01, kb10, kb11;
    {
      const _Float16* kr0 = kk + (size_t)(j0 + ln) * 64;
      const _Float16* kr1 = kr0 + 16 * 64;
      kb00.q[0] = *(const uint4*)(kr0 + hl * 8);
      kb00.q[1] = *(const uint4*)(kr0 + 16 + hl * 8);
      kb01.q[0] = *(const uint4*)(kr0 + 32 + hl * 8);
      kb01.q[1] = *(const uint4*)(kr0 + 48 + hl * 8);
      kb10.q[0] = *(const uint4*)(kr1 + hl * 8);
      kb10.q[1] = *(const uint4*)(kr1 + 16 + hl * 8);
      kb11.q[0] = *(const uint4*)(kr1 + 32 + hl * 8);
      kb11.q[1] = *(const uint4*)(kr1 + 48 + hl * 8);
    }
    v8f s0 = {}, s1 = {};
    s0 = wmma16(qa0, kb00, s0);   // S = Q(16x64) * K^T(64x32)
    s0 = wmma16(qa1, kb01, s0);
    s1 = wmma16(qa0, kb10, s1);
    s1 = wmma16(qa1, kb11, s1);

    const bool v0 = (j0 + ln) < Mk;
    const bool v1 = (j0 + 16 + ln) < Mk;
#pragma unroll
    for (int r = 0; r < 8; ++r) {
      float a = v0 ? s0[r] * scale : -__builtin_inff();
      float b = v1 ? s1[r] * scale : -__builtin_inff();
      float t = fmaxf(a, b);
      t = fmaxf(t, SWZ_XOR(t, 0x041f));
      t = fmaxf(t, SWZ_XOR(t, 0x081f));
      t = fmaxf(t, SWZ_XOR(t, 0x101f));
      t = fmaxf(t, SWZ_XOR(t, 0x201f));
      const float nm = fmaxf(mr[r], t);
      float alpha, p0, p1;
      if (nm == -__builtin_inff()) { alpha = 1.f; p0 = 0.f; p1 = 0.f; }
      else { alpha = __expf(mr[r] - nm); p0 = __expf(a - nm); p1 = __expf(b - nm); }
      float ps = p0 + p1;
      ps += SWZ_XOR(ps, 0x041f);
      ps += SWZ_XOR(ps, 0x081f);
      ps += SWZ_XOR(ps, 0x101f);
      ps += SWZ_XOR(ps, 0x201f);
      lr[r] = lr[r] * alpha + ps;
      mr[r] = nm;
      o0[r] *= alpha; o1[r] *= alpha; o2[r] *= alpha;
      Pl[wid][hl * 8 + r][ln]      = (_Float16)p0;   // wave-private, DS in-order
      Pl[wid][hl * 8 + r][16 + ln] = (_Float16)p1;
    }

    FragH pf, vf0, vf1, vf2;
    pf.q[0] = *(const uint4*)(&Pl[wid][ln][hl * 8]);
    pf.q[1] = *(const uint4*)(&Pl[wid][ln][16 + hl * 8]);
    {
      const _Float16* vr0 = vt + (size_t)(ln) * Mkp + j0;
      const _Float16* vr1 = vt + (size_t)(16 + ln) * Mkp + j0;
      const _Float16* vr2 = vt + (size_t)(32 + ln) * Mkp + j0;
      vf0.q[0] = *(const uint4*)(vr0 + hl * 8);
      vf0.q[1] = *(const uint4*)(vr0 + 16 + hl * 8);
      vf1.q[0] = *(const uint4*)(vr1 + hl * 8);
      vf1.q[1] = *(const uint4*)(vr1 + 16 + hl * 8);
      vf2.q[0] = *(const uint4*)(vr2 + hl * 8);
      vf2.q[1] = *(const uint4*)(vr2 + 16 + hl * 8);
    }
    o0 = wmma16(pf, vf0, o0);     // O += P(16x32) * V(32x48)
    o1 = wmma16(pf, vf1, o1);
    o2 = wmma16(pf, vf2, o2);
  }

#pragma unroll
  for (int r = 0; r < 8; ++r) {
    const float rl = (lr[r] > 0.f) ? (1.f / lr[r]) : 0.f;
    _Float16* ob = out + ((size_t)bg * Nq + qrow0 + hl * 8 + r) * 320 + hh * 40;
    ob[ln]      = (_Float16)(o0[r] * rl);
    ob[16 + ln] = (_Float16)(o1[r] * rl);
    if (ln < 8) ob[32 + ln] = (_Float16)(o2[r] * rl);  // d=32..39 only
  }
}

// ---------------------------------------------------------------------------
// Helpers
// ---------------------------------------------------------------------------
__global__ __launch_bounds__(256) void layernorm_f16_k(
    const float* __restrict__ x, const float* __restrict__ gw,
    const float* __restrict__ bw, _Float16* __restrict__ out, int rows)
{
  const int wid = threadIdx.x >> 5, lane = threadIdx.x & 31;
  const int row = blockIdx.x * 8 + wid;
  if (row >= rows) return;
  const float* xr = x + (size_t)row * 320;
  float vals[10], s = 0.f, s2 = 0.f;
#pragma unroll
  for (int t = 0; t < 10; ++t) {
    float v = xr[lane + 32 * t];
    vals[t] = v; s += v; s2 += v * v;
  }
  s += SWZ_XOR(s, 0x041f); s2 += SWZ_XOR(s2, 0x041f);
  s += SWZ_XOR(s, 0x081f); s2 += SWZ_XOR(s2, 0x081f);
  s += SWZ_XOR(s, 0x101f); s2 += SWZ_XOR(s2, 0x101f);
  s += SWZ_XOR(s, 0x201f); s2 += SWZ_XOR(s2, 0x201f);
  s += SWZ_XOR(s, 0x401f); s2 += SWZ_XOR(s2, 0x401f);
  const float mean = s * (1.f / 320.f);
  const float var  = s2 * (1.f / 320.f) - mean * mean;
  const float rstd = rsqrtf(var + 1e-5f);
  _Float16* orow = out + (size_t)row * 320;
#pragma unroll
  for (int t = 0; t < 10; ++t) {
    const int c = lane + 32 * t;
    orow[c] = (_Float16)((vals[t] - mean) * rstd * gw[c] + bw[c]);
  }
}

__global__ void cvt_f16_k(const float* __restrict__ in, _Float16* __restrict__ out, int n) {
  const int i = blockIdx.x * 256 + threadIdx.x;
  if (i < n) out[i] = (_Float16)in[i];
}

__global__ void zero_u4_k(uint4* __restrict__ p, int n4) {
  const int i = blockIdx.x * 256 + threadIdx.x;
  if (i < n4) { uint4 z = {0u, 0u, 0u, 0u}; p[i] = z; }
}

__global__ void geglu_k(const _Float16* __restrict__ h, _Float16* __restrict__ o, int total) {
  const int i = blockIdx.x * 256 + threadIdx.x;
  if (i >= total) return;
  const int r = i / 1280, c = i - r * 1280;
  const float a = (float)h[(size_t)r * 2560 + c];
  const float g = (float)h[(size_t)r * 2560 + 1280 + c];
  const float ge = 0.5f * g * (1.f + erff(g * 0.70710678118654752f));  // exact gelu
  o[i] = (_Float16)(a * ge);
}

__global__ void combine_k(const float* __restrict__ g, const float* __restrict__ gsp,
                          const float* __restrict__ x1, const float* __restrict__ coef,
                          const unsigned char* __restrict__ masks, float* __restrict__ x2)
{
  const int i = blockIdx.x * 256 + threadIdx.x;  // over N*C = 4096*320
  if (i >= 4096 * 320) return;
  const int tok = i / 320;
  const float g0 = g[i];
  x2[i] = g0 + x1[i];                                    // batch 0
  float v = g[4096 * 320 + i];                           // batch 1
#pragma unroll
  for (int o = 0; o < 4; ++o) {
    if (masks[o * 4096 + tok])
      v += coef[o] * (gsp[(size_t)o * 4096 * 320 + i] - g0);
  }
  x2[4096 * 320 + i] = v + x1[4096 * 320 + i];
}

// ---------------------------------------------------------------------------
extern "C" void kernel_launch(void* const* d_in, const int* in_sizes, int n_in,
                              void* d_out, int out_size, void* d_ws, size_t ws_size,
                              hipStream_t stream)
{
  (void)in_sizes; (void)n_in; (void)out_size; (void)ws_size;
  const float* x    = (const float*)d_in[0];
  const float* ctx  = (const float*)d_in[1];
  const float* ccs  = (const float*)d_in[2];
  const float* coef = (const float*)d_in[3];
  const unsigned char* masks = (const unsigned char*)d_in[4];  // bool (1 byte)
  const float* a1wq = (const float*)d_in[5];
  const float* a1wk = (const float*)d_in[6];
  const float* a1wv = (const float*)d_in[7];
  const float* a1wo = (const float*)d_in[8];
  const float* a1bo = (const float*)d_in[9];
  const float* a2wq = (const float*)d_in[10];
  const float* a2wk = (const float*)d_in[11];
  const float* a2wv = (const float*)d_in[12];
  const float* a2wo = (const float*)d_in[13];
  const float* a2bo = (const float*)d_in[14];
  const float* ln1g = (const float*)d_in[15];
  const float* ln1b = (const float*)d_in[16];
  const float* ln2g = (const float*)d_in[17];
  const float* ln2b = (const float*)d_in[18];
  const float* ln3g = (const float*)d_in[19];
  const float* ln3b = (const float*)d_in[20];
  const float* ffw1 = (const float*)d_in[21];
  const float* ffb1 = (const float*)d_in[22];
  const float* ffw2 = (const float*)d_in[23];
  const float* ffb2 = (const float*)d_in[24];

  char* base = (char*)d_ws;
  size_t off = 0;
  auto alloc = [&](size_t bytes) -> void* {
    void* p = base + off;
    off += (bytes + 255) & ~(size_t)255;
    return p;
  };
  const float SCALE = 0.158113883008419f;  // 40^-0.5

  // f16 weights / contexts
  _Float16* w1q = (_Float16*)alloc((size_t)320 * 320 * 2);
  _Float16* w1k = (_Float16*)alloc((size_t)320 * 320 * 2);
  _Float16* w1v = (_Float16*)alloc((size_t)320 * 320 * 2);
  _Float16* w1o = (_Float16*)alloc((size_t)320 * 320 * 2);
  _Float16* w2q = (_Float16*)alloc((size_t)320 * 320 * 2);
  _Float16* w2k = (_Float16*)alloc((size_t)768 * 320 * 2);
  _Float16* w2v = (_Float16*)alloc((size_t)768 * 320 * 2);
  _Float16* w2o = (_Float16*)alloc((size_t)320 * 320 * 2);
  _Float16* wf1 = (_Float16*)alloc((size_t)320 * 2560 * 2);
  _Float16* wf2 = (_Float16*)alloc((size_t)1280 * 320 * 2);
  _Float16* ctx16 = (_Float16*)alloc((size_t)154 * 768 * 2);
  _Float16* ccs16 = (_Float16*)alloc((size_t)616 * 768 * 2);
  // activations
  _Float16* xn    = (_Float16*)alloc((size_t)8192 * 320 * 2);     // LN output (reused 3x)
  _Float16* qbuf  = (_Float16*)alloc((size_t)16 * 4096 * 64 * 2); // [16,4096,64]
  _Float16* kbuf  = (_Float16*)alloc((size_t)16 * 4096 * 64 * 2);
  _Float16* vtbuf = (_Float16*)alloc((size_t)16 * 64 * 4096 * 2); // [16,64,4096]
  _Float16* aobuf = (_Float16*)alloc((size_t)8192 * 320 * 2);     // attn out (reused)
  _Float16* gsatt = (_Float16*)alloc((size_t)4 * 4096 * 320 * 2); // object attn out (b=1)
  _Float16* kctx  = (_Float16*)alloc((size_t)16 * 96 * 64 * 2);
  _Float16* vtctx = (_Float16*)alloc((size_t)16 * 64 * 96 * 2);
  _Float16* kobj  = (_Float16*)alloc((size_t)64 * 96 * 64 * 2);
  _Float16* vtobj = (_Float16*)alloc((size_t)64 * 64 * 96 * 2);
  _Float16* hbuf  = (_Float16*)alloc((size_t)8192 * 2560 * 2);    // FF1 out
  _Float16* ggbuf = (_Float16*)alloc((size_t)8192 * 1280 * 2);    // GEGLU out
  float* x1buf = (float*)alloc((size_t)8192 * 320 * 4);
  float* gf    = (float*)alloc((size_t)8192 * 320 * 4);
  float* gsp   = (float*)alloc((size_t)4 * 4096 * 320 * 4);
  float* x2buf = (float*)alloc((size_t)8192 * 320 * 4);

  auto cvt = [&](const float* s, _Float16* d, int n) {
    cvt_f16_k<<<(n + 255) / 256, 256, 0, stream>>>(s, d, n);
  };
  cvt(a1wq, w1q, 320 * 320);  cvt(a1wk, w1k, 320 * 320);
  cvt(a1wv, w1v, 320 * 320);  cvt(a1wo, w1o, 320 * 320);
  cvt(a2wq, w2q, 320 * 320);  cvt(a2wk, w2k, 768 * 320);
  cvt(a2wv, w2v, 768 * 320);  cvt(a2wo, w2o, 320 * 320);
  cvt(ffw1, wf1, 320 * 2560); cvt(ffw2, wf2, 1280 * 320);
  cvt(ctx, ctx16, 154 * 768); cvt(ccs, ccs16, 616 * 768);

  // zero Q buffer once per call so the d=40..63 pad is exactly 0
  {
    const int n4 = (16 * 4096 * 64 * 2) / 16;
    zero_u4_k<<<(n4 + 255) / 256, 256, 0, stream>>>((uint4*)qbuf, n4);
  }

  // ---- self attention ----
  layernorm_f16_k<<<1024, 256, 0, stream>>>(x, ln1g, ln1b, xn, 8192);
  gemm_f16_wmma<1><<<dim3(5, 128), 256, 0, stream>>>(xn, w1q, nullptr, nullptr, qbuf, 8192, 320, 320, 4096, 4096);
  gemm_f16_wmma<1><<<dim3(5, 128), 256, 0, stream>>>(xn, w1k, nullptr, nullptr, kbuf, 8192, 320, 320, 4096, 4096);
  gemm_f16_wmma<2><<<dim3(5, 128), 256, 0, stream>>>(xn, w1v, nullptr, nullptr, vtbuf, 8192, 320, 320, 4096, 4096);
  flash_attn_wmma<<<dim3(32, 16), 256, 0, stream>>>(qbuf, kbuf, vtbuf, aobuf, 4096, 4096, 4096, SCALE);
  gemm_f16_wmma<4><<<dim3(5, 128), 256, 0, stream>>>(aobuf, w1o, a1bo, x, x1buf, 8192, 320, 320, 0, 1);

  // ---- cross attention (global + per-object) ----
  layernorm_f16_k<<<1024, 256, 0, stream>>>(x1buf, ln2g, ln2b, xn, 8192);
  gemm_f16_wmma<1><<<dim3(5, 128), 256, 0, stream>>>(xn, w2q, nullptr, nullptr, qbuf, 8192, 320, 320, 4096, 4096);
  gemm_f16_wmma<1><<<dim3(5, 3),  256, 0, stream>>>(ctx16, w2k, nullptr, nullptr, kctx, 154, 320, 768, 77, 96);
  gemm_f16_wmma<2><<<dim3(5, 3),  256, 0, stream>>>(ctx16, w2v, nullptr, nullptr, vtctx, 154, 320, 768, 77, 96);
  gemm_f16_wmma<1><<<dim3(5, 10), 256, 0, stream>>>(ccs16, w2k, nullptr, nullptr, kobj, 616, 320, 768, 77, 96);
  gemm_f16_wmma<2><<<dim3(5, 10), 256, 0, stream>>>(ccs16, w2v, nullptr, nullptr, vtobj, 616, 320, 768, 77, 96);
  flash_attn_wmma<<<dim3(32, 16), 256, 0, stream>>>(qbuf, kctx, vtctx, aobuf, 4096, 77, 96, SCALE);
  for (int o = 0; o < 4; ++o) {
    const size_t goff = (size_t)((o * 2 + 1) * 8);
    flash_attn_wmma<<<dim3(32, 8), 256, 0, stream>>>(
        qbuf + (size_t)8 * 4096 * 64,          // batch-1 head groups
        kobj + goff * 96 * 64, vtobj + goff * 64 * 96,
        gsatt + (size_t)o * 4096 * 320, 4096, 77, 96, SCALE);
  }
  gemm_f16_wmma<3><<<dim3(5, 128), 256, 0, stream>>>(aobuf, w2o, a2bo, nullptr, gf, 8192, 320, 320, 0, 1);
  gemm_f16_wmma<3><<<dim3(5, 256), 256, 0, stream>>>(gsatt, w2o, a2bo, nullptr, gsp, 16384, 320, 320, 0, 1);
  combine_k<<<(4096 * 320 + 255) / 256, 256, 0, stream>>>(gf, gsp, x1buf, coef, masks, x2buf);

  // ---- GEGLU feed-forward ----
  layernorm_f16_k<<<1024, 256, 0, stream>>>(x2buf, ln3g, ln3b, xn, 8192);
  gemm_f16_wmma<0><<<dim3(40, 128), 256, 0, stream>>>(xn, wf1, ffb1, nullptr, hbuf, 8192, 2560, 320, 0, 1);
  geglu_k<<<(8192 * 1280 + 255) / 256, 256, 0, stream>>>(hbuf, ggbuf, 8192 * 1280);
  gemm_f16_wmma<4><<<dim3(5, 128), 256, 0, stream>>>(ggbuf, wf2, ffb2, x2buf, d_out, 8192, 320, 1280, 0, 1);
}
